// EGNN_1958505087691
// MI455X (gfx1250) — compile-verified, matrix operand order checked by reading, not run on hardware
//
#include <hip/hip_runtime.h>
#include <hip/hip_bf16.h>
#include <stdint.h>

// ---------------------------------------------------------------------------
// EGNN on MI455X (gfx1250): wave32 + v_wmma_f32_16x16x32_f16
// Persistent edge kernel: grid-stride over 16-edge tiles, all weight
// B-fragments register-resident for the lifetime of the block.
// ---------------------------------------------------------------------------

typedef __attribute__((ext_vector_type(16))) _Float16 v16h;
typedef __attribute__((ext_vector_type(8)))  _Float16 v8h;
typedef __attribute__((ext_vector_type(8)))  float    v8f;

union AFrag { v16h v; v8h h2[2]; };

__device__ __forceinline__ v8f wmma16(v16h a, v16h b, v8f c) {
  // D = A(16x32 f16) * B(32x16 f16) + C(16x16 f32)
  return __builtin_amdgcn_wmma_f32_16x16x32_f16(
      /*neg_a=*/false, a, /*neg_b=*/false, b,
      /*c_mod=*/(short)0, c, /*reuse_a=*/false, /*reuse_b=*/false);
}

// A-fragment from an f16 row pointer. Layout (16-bit A 16x32):
//   lane<16: K = k0+{0..7} and k0+{16..23};  lane>=16: +8 on both chunks.
__device__ __forceinline__ v16h load_a_f16(const _Float16* rowp, int k0, int kb) {
  AFrag a;
  a.h2[0] = *(const v8h*)(rowp + k0 + kb);
  a.h2[1] = *(const v8h*)(rowp + k0 + kb + 16);
  return a.v;
}

// Same, but source row is f32 (converted on the fly).
__device__ __forceinline__ v16h load_a_f32(const float* rowp, int k0, int kb) {
  AFrag a;
  const float* p = rowp + k0 + kb;
  #pragma unroll
  for (int i = 0; i < 8; ++i) {
    a.h2[0][i] = (_Float16)p[i];
    a.h2[1][i] = (_Float16)p[16 + i];
  }
  return a.v;
}

// B-fragment: weights pre-shuffled so each lane reads 16 contiguous f16 (32B).
__device__ __forceinline__ v16h load_b(const _Float16* wp, int kt, int wv, int lane) {
  return *(const v16h*)(wp + (((size_t)kt * 8 + wv) * 32 + lane) * 16);
}

// ---------------------------------------------------------------------------
// Weight pre-shuffle: src [K,128] f32 row-major -> B-fragment order (f16).
// ---------------------------------------------------------------------------
__global__ void prep_w_kernel(const float* __restrict__ src,
                              _Float16* __restrict__ dst, int K) {
  int t = blockIdx.x * 256 + threadIdx.x;
  if (t >= K * 128) return;
  int i    = t & 15;
  int lane = (t >> 4) & 31;
  int nt   = (t >> 9) & 7;
  int kt   = t >> 12;
  int kb   = (lane & 16) ? 8 : 0;
  int k    = kt * 32 + kb + ((i < 8) ? i : i + 8);
  int n    = nt * 16 + (lane & 15);
  dst[t] = (_Float16)src[(size_t)k * 128 + n];
}

// h0 = h_in @ emb_W + emb_b  (N x 16 @ 16 x 128; tiny, VALU is fine)
__global__ void embed_kernel(const float* __restrict__ h_in,
                             const float* __restrict__ W,
                             const float* __restrict__ b,
                             float* __restrict__ h_f32,
                             _Float16* __restrict__ h16, int N) {
  int t = blockIdx.x * 256 + threadIdx.x;
  if (t >= N * 128) return;
  int node = t >> 7, c = t & 127;
  float s = b[c];
  const float* hp = h_in + (size_t)node * 16;
  #pragma unroll
  for (int p = 0; p < 16; ++p) s += hp[p] * W[p * 128 + c];
  h_f32[t] = s;
  h16[t]   = (_Float16)s;
}

__global__ void fzero_kernel(float* __restrict__ p, int n) {
  int t = blockIdx.x * 256 + threadIdx.x;
  if (t < n) p[t] = 0.0f;
}
__global__ void fcopy_kernel(float* __restrict__ d, const float* __restrict__ s, int n) {
  int t = blockIdx.x * 256 + threadIdx.x;
  if (t < n) d[t] = s[t];
}

// ---------------------------------------------------------------------------
// Persistent edge kernel. Per 16-edge tile:
//   z  = relu(h[row] @ eW1a + h[col] @ eW1b + rel_dist (x) eW1[256,:] + eb1)
//   e  = z @ eW2 + eb2                 -> atomic segment-sum into m[row]
//   t  = relu(e @ cW1 + cb1)
//   a  = t . cW2 + cb2                 -> x_next[row] += a * rel_pos (atomic)
// B-fragments (8 + 4 + 4 v16h = 128 VGPRs) live in registers across tiles.
// ---------------------------------------------------------------------------
__global__ __launch_bounds__(256) void edge_kernel(
    const _Float16* __restrict__ h16, const float* __restrict__ x_cur,
    float* __restrict__ x_next, float* __restrict__ m_out,
    const int* __restrict__ ei,
    const _Float16* __restrict__ eW1p, const float* __restrict__ eW1_last,
    const float* __restrict__ eb1,
    const _Float16* __restrict__ eW2p, const float* __restrict__ eb2,
    const _Float16* __restrict__ cW1p, const float* __restrict__ cb1,
    const float* __restrict__ cW2, const float* __restrict__ cb2,
    int E, int numTiles) {
  __shared__ _Float16 s_z[16][128];
  __shared__ _Float16 s_e[16][128];
  __shared__ _Float16 s_t[16][128];
  __shared__ float s_cw2[128];
  __shared__ int   s_row[16], s_col[16];
  __shared__ float s_rp[16][3], s_rd[16];

  const int tid  = threadIdx.x;
  const int lane = tid & 31;
  const int wv   = tid >> 5;

  const int nloc = lane & 15;
  const int n    = wv * 16 + nloc;
  const int hi8  = (lane & 16) ? 8 : 0;
  const int kb   = hi8;
  const int mrow = nloc;

  // ---- block-lifetime state: weight fragments + per-lane scalars ----
  v16h B1[8], B2[4], B3[4];
  #pragma unroll
  for (int kt = 0; kt < 8; ++kt) B1[kt] = load_b(eW1p, kt, wv, lane);
  #pragma unroll
  for (int kt = 0; kt < 4; ++kt) B2[kt] = load_b(eW2p, kt, wv, lane);
  #pragma unroll
  for (int kt = 0; kt < 4; ++kt) B3[kt] = load_b(cW1p, kt, wv, lane);
  const float bias1 = eb1[n];
  const float w256  = eW1_last[n];
  const float bias2 = eb2[n];
  const float bias3 = cb1[n];
  const float cb2v  = cb2[0];
  if (tid < 128) s_cw2[tid] = cW2[tid];

  const int tileStride = gridDim.x;

  for (int tile = blockIdx.x; tile < numTiles; tile += tileStride) {
    const int e0 = tile * 16;

    if (tid < 16) {
      int r = 0, c = 0;
      float d = 0.0f;
      if (e0 + tid < E) {
        r = ei[e0 + tid];
        c = ei[E + e0 + tid];
        #pragma unroll
        for (int k = 0; k < 3; ++k) {
          float rp = x_cur[(size_t)r * 3 + k] - x_cur[(size_t)c * 3 + k];
          s_rp[tid][k] = rp;
          d += rp * rp;
        }
      } else {
        s_rp[tid][0] = s_rp[tid][1] = s_rp[tid][2] = 0.0f;
      }
      s_row[tid] = r; s_col[tid] = c; s_rd[tid] = d;
      // Prefetch next tile's indices (gfx1250 global_prefetch_b8).
      int enx = e0 + tileStride * 16 + tid;
      if (enx < E) {
        __builtin_prefetch(&ei[enx], 0, 1);
        __builtin_prefetch(&ei[E + enx], 0, 1);
      }
    }
    __syncthreads();

    const _Float16* hr = h16 + (size_t)s_row[mrow] * 128;
    const _Float16* hc = h16 + (size_t)s_col[mrow] * 128;

    // ---- GEMM 1: K=256 (h_row | h_col) + rank-1 rel_dist term ----
    v8f acc;
    #pragma unroll
    for (int j = 0; j < 8; ++j) acc[j] = bias1 + s_rd[j + hi8] * w256;
    #pragma unroll
    for (int kt = 0; kt < 4; ++kt) {
      acc = wmma16(load_a_f16(hr, kt * 32, kb), B1[kt],     acc);
      acc = wmma16(load_a_f16(hc, kt * 32, kb), B1[kt + 4], acc);
    }
    #pragma unroll
    for (int j = 0; j < 8; ++j) {
      float z = acc[j] > 0.0f ? acc[j] : 0.0f;
      s_z[j + hi8][n] = (_Float16)z;
    }
    __syncthreads();

    // ---- GEMM 2: e = z @ eW2 + eb2 ----
    v8f acc2;
    #pragma unroll
    for (int j = 0; j < 8; ++j) acc2[j] = bias2;
    #pragma unroll
    for (int kt = 0; kt < 4; ++kt)
      acc2 = wmma16(load_a_f16(&s_z[mrow][0], kt * 32, kb), B2[kt], acc2);

    #pragma unroll
    for (int j = 0; j < 8; ++j) {
      int mj = j + hi8;
      if (e0 + mj < E)
        atomicAdd(&m_out[(size_t)s_row[mj] * 128 + n], acc2[j]);
      s_e[mj][n] = (_Float16)acc2[j];
    }
    __syncthreads();

    // ---- GEMM 3: t = relu(e @ cW1 + cb1) ----
    v8f acc3;
    #pragma unroll
    for (int j = 0; j < 8; ++j) acc3[j] = bias3;
    #pragma unroll
    for (int kt = 0; kt < 4; ++kt)
      acc3 = wmma16(load_a_f16(&s_e[mrow][0], kt * 32, kb), B3[kt], acc3);
    #pragma unroll
    for (int j = 0; j < 8; ++j) {
      float t = acc3[j] > 0.0f ? acc3[j] : 0.0f;
      s_t[j + hi8][n] = (_Float16)t;
    }
    __syncthreads();

    // ---- alpha = t . cW2 + cb2 ; x_next[row] += alpha * rel_pos ----
    if (tid < 16 && e0 + tid < E) {
      float dot = cb2v;
      for (int k = 0; k < 128; ++k) dot += (float)s_t[tid][k] * s_cw2[k];
      int r = s_row[tid];
      #pragma unroll
      for (int d = 0; d < 3; ++d)
        atomicAdd(&x_next[(size_t)r * 3 + d], dot * s_rp[tid][d]);
    }
    // No trailing barrier needed: only wave 0 rewrites s_row/s_rp/s_rd (after
    // its own alpha step), and other waves touch them only after the next
    // iteration's first __syncthreads().
  }
}

// ---------------------------------------------------------------------------
// Node kernel: h += MLP(concat(h, m)) in place (each block owns its 16 rows)
// ---------------------------------------------------------------------------
__global__ __launch_bounds__(256) void node_kernel(
    float* __restrict__ h_f32, _Float16* __restrict__ h16,
    const float* __restrict__ m_in,
    const _Float16* __restrict__ nW1p, const float* __restrict__ nb1,
    const _Float16* __restrict__ nW2p, const float* __restrict__ nb2, int N) {
  __shared__ _Float16 s_z[16][128];

  const int tid  = threadIdx.x;
  const int lane = tid & 31;
  const int wv   = tid >> 5;
  const int nb0  = blockIdx.x * 16;
  const int nloc = lane & 15;
  const int n    = wv * 16 + nloc;
  const int hi8  = (lane & 16) ? 8 : 0;
  const int kb   = hi8;

  int rowh = nb0 + nloc; if (rowh >= N) rowh = N - 1;
  const _Float16* hr = h16  + (size_t)rowh * 128;
  const float*    mr = m_in + (size_t)rowh * 128;

  v8f acc;
  { const float b1 = nb1[n];
    #pragma unroll
    for (int j = 0; j < 8; ++j) acc[j] = b1; }
  #pragma unroll
  for (int kt = 0; kt < 4; ++kt) {
    acc = wmma16(load_a_f16(hr, kt * 32, kb), load_b(nW1p, kt,     wv, lane), acc);
    acc = wmma16(load_a_f32(mr, kt * 32, kb), load_b(nW1p, kt + 4, wv, lane), acc);
  }
  #pragma unroll
  for (int j = 0; j < 8; ++j) {
    float z = acc[j] > 0.0f ? acc[j] : 0.0f;
    s_z[j + hi8][n] = (_Float16)z;
  }
  __syncthreads();  // all h16 reads complete before in-place writes below

  v8f acc2;
  { const float b2 = nb2[n];
    #pragma unroll
    for (int j = 0; j < 8; ++j) acc2[j] = b2; }
  #pragma unroll
  for (int kt = 0; kt < 4; ++kt)
    acc2 = wmma16(load_a_f16(&s_z[nloc][0], kt * 32, kb), load_b(nW2p, kt, wv, lane), acc2);

  #pragma unroll
  for (int j = 0; j < 8; ++j) {
    int r = nb0 + j + hi8;
    if (r < N) {
      size_t idx = (size_t)r * 128 + n;
      float hn = h_f32[idx] + acc2[j];
      h_f32[idx] = hn;
      h16[idx]   = (_Float16)hn;
    }
  }
}

// ---------------------------------------------------------------------------
extern "C" void kernel_launch(void* const* d_in, const int* in_sizes, int n_in,
                              void* d_out, int out_size, void* d_ws, size_t ws_size,
                              hipStream_t stream) {
  const float* h_in  = (const float*)d_in[0];
  const float* x_in  = (const float*)d_in[1];
  const int*   ei    = (const int*)  d_in[2];
  const float* emb_W = (const float*)d_in[3];
  const float* emb_b = (const float*)d_in[4];
  const float* eW1   = (const float*)d_in[5];
  const float* eb1   = (const float*)d_in[6];
  const float* eW2   = (const float*)d_in[7];
  const float* eb2   = (const float*)d_in[8];
  const float* nW1   = (const float*)d_in[9];
  const float* nb1   = (const float*)d_in[10];
  const float* nW2   = (const float*)d_in[11];
  const float* nb2   = (const float*)d_in[12];
  const float* cW1   = (const float*)d_in[13];
  const float* cb1   = (const float*)d_in[14];
  const float* cW2   = (const float*)d_in[15];
  const float* cb2   = (const float*)d_in[16];

  const int N = in_sizes[0] / 16;   // h is [N,16]
  const int E = in_sizes[2] / 2;    // edge_index is [2,E]

  char* w = (char*)d_ws;
  auto alloc = [&](size_t bytes) -> char* {
    char* p = w; w += (bytes + 255) & ~(size_t)255; return p;
  };
  float*    hf   = (float*)   alloc((size_t)N * 128 * 4);
  _Float16* h16  = (_Float16*)alloc((size_t)N * 128 * 2);
  float*    mbuf = (float*)   alloc((size_t)N * 128 * 4);
  float*    xA   = (float*)   alloc((size_t)N * 3 * 4);
  float*    xB   = (float*)   alloc((size_t)N * 3 * 4);
  _Float16* eW1p = (_Float16*)alloc((size_t)256 * 128 * 2);
  _Float16* eW2p = (_Float16*)alloc((size_t)128 * 128 * 2);
  _Float16* nW1p = (_Float16*)alloc((size_t)256 * 128 * 2);
  _Float16* nW2p = (_Float16*)alloc((size_t)128 * 128 * 2);
  _Float16* cW1p = (_Float16*)alloc((size_t)128 * 128 * 2);
  (void)n_in; (void)out_size; (void)ws_size;

  const int T = 256;
  auto g = [](int n, int t) { return (n + t - 1) / t; };

  // Pre-shuffle weights into WMMA B-fragment order (f16).
  prep_w_kernel<<<g(256 * 128, T), T, 0, stream>>>(eW1, eW1p, 256);
  prep_w_kernel<<<g(128 * 128, T), T, 0, stream>>>(eW2, eW2p, 128);
  prep_w_kernel<<<g(256 * 128, T), T, 0, stream>>>(nW1, nW1p, 256);
  prep_w_kernel<<<g(128 * 128, T), T, 0, stream>>>(nW2, nW2p, 128);
  prep_w_kernel<<<g(128 * 128, T), T, 0, stream>>>(cW1, cW1p, 128);

  // Embedding + x init.
  embed_kernel<<<g(N * 128, T), T, 0, stream>>>(h_in, emb_W, emb_b, hf, h16, N);
  fcopy_kernel<<<g(N * 3, T), T, 0, stream>>>(xA, x_in, N * 3);

  const int eTiles  = g(E, 16);
  const int eBlocks = eTiles < 4096 ? eTiles : 4096;  // persistent blocks
  const int nBlocks = g(N, 16);
  const float* eW1_last = eW1 + (size_t)256 * 128;    // rel_dist row (f32)

  // ---- Layer 1: read xA, accumulate into xB ----
  fzero_kernel<<<g(N * 128, T), T, 0, stream>>>(mbuf, N * 128);
  fcopy_kernel<<<g(N * 3, T), T, 0, stream>>>(xB, xA, N * 3);
  edge_kernel<<<eBlocks, T, 0, stream>>>(h16, xA, xB, mbuf, ei,
                                         eW1p, eW1_last, eb1, eW2p, eb2,
                                         cW1p, cb1, cW2, cb2, E, eTiles);
  node_kernel<<<nBlocks, T, 0, stream>>>(hf, h16, mbuf, nW1p, nb1, nW2p, nb2, N);

  // ---- Layer 2: read xB, accumulate into xA ----
  fzero_kernel<<<g(N * 128, T), T, 0, stream>>>(mbuf, N * 128);
  fcopy_kernel<<<g(N * 3, T), T, 0, stream>>>(xA, xB, N * 3);
  edge_kernel<<<eBlocks, T, 0, stream>>>(h16, xB, xA, mbuf, ei,
                                         eW1p, eW1_last, eb1, eW2p, eb2,
                                         cW1p, cb1, cW2, cb2, E, eTiles);
  node_kernel<<<nBlocks, T, 0, stream>>>(hf, h16, mbuf, nW1p, nb1, nW2p, nb2, N);

  // ---- Output: [h (N*128) | x (N*3)] ----
  float* out = (float*)d_out;
  fcopy_kernel<<<g(N * 128, T), T, 0, stream>>>(out, hf, N * 128);
  fcopy_kernel<<<g(N * 3, T), T, 0, stream>>>(out + (size_t)N * 128, xA, N * 3);
}
